// TrajConv_11802570130226
// MI455X (gfx1250) — compile-verified, matrix operand order checked by reading, not run on hardware
//
#include <hip/hip_runtime.h>

typedef __attribute__((ext_vector_type(16))) _Float16     v16h;
typedef __attribute__((ext_vector_type(8)))  float        v8f;
typedef __attribute__((ext_vector_type(8)))  unsigned int v8u;
typedef __attribute__((ext_vector_type(4)))  unsigned int v4u;
typedef __attribute__((ext_vector_type(8)))  int          v8i;
typedef __attribute__((ext_vector_type(4)))  int          v4i;

#define CIN   32
#define COUT  64
#define KTAP  27
#define KDIM  (CIN * KTAP)      // 864 (reduction dim: c*27 + tap)
#define TT    8
#define HH    56
#define WW    56
#define SPW   (TT * HH * WW)    // 25088 spatial elements per (n, channel)
#define NPOS  (2 * SPW)         // 50176 total output positions
#define POS_PER_BLK 32
#define OFFCH (3 * KTAP)        // 81 offset channels

__global__ __launch_bounds__(256)
void deform_conv3d_wmma(const float* __restrict__ x,
                        const float* __restrict__ off,
                        const float* __restrict__ wgt,
                        const float* __restrict__ bias,
                        float* __restrict__ out) {
    // col tile: 32 pos x 864 K (f16)  = 55296 B
    // off tile: 81 ch  x 32 pos (f32) = 10368 B   (TDM destination)
    __shared__ __align__(16) _Float16 colLDS[POS_PER_BLK * KDIM];
    __shared__ __align__(16) float    offLDS[OFFCH * POS_PER_BLK];

    const int tid  = threadIdx.x;
    const int wave = tid >> 5;
    const int lane = tid & 31;
    const int p0   = blockIdx.x * POS_PER_BLK;      // grid = NPOS/32 exactly
    const int nb0  = p0 / SPW;                      // whole block in one batch
    const int rem0 = p0 - nb0 * SPW;                // (SPW % 32 == 0)

    // ------- Phase 0: TDM load of the 81x32 offset tile into LDS -----------
    if (wave == 0) {
        // 57-bit global byte address of tile start: offset[nb0, 0, rem0]
        const unsigned long long ga =
            (unsigned long long)(uintptr_t)off +
            4ull * ((unsigned long long)nb0 * OFFCH * SPW + (unsigned long long)rem0);
        // LDS aperture keeps the raw LDS byte offset in addr[31:0]
        const unsigned int ldsoff = (unsigned int)(uintptr_t)&offLDS[0];

        // D# group 0: count=1 | lds_addr | global_addr[56:0] | type=2
        v4u g0;
        g0[0] = 1u;                                       // count = 1, user mode
        g0[1] = ldsoff;                                   // lds_addr
        g0[2] = (unsigned int)(ga & 0xFFFFFFFFull);       // global_addr[31:0]
        g0[3] = (unsigned int)((ga >> 32) & 0x1FFFFFFull) // global_addr[56:32]
                | 0x80000000u;                            // type = 2 (bits 127:126 = 10b)

        // D# group 1: data_size=4B; tensor 32x81; tile 32x81; dim0 stride 25088
        v8i g1;
        g1[0] = 0x00020000;            // data_size = 2 (4 bytes), mask/pad = 0
        g1[1] = (POS_PER_BLK << 16);   // tensor_dim0[15:0] = 32
        g1[2] = (OFFCH << 16);         // tensor_dim0[31:16]=0 | tensor_dim1[15:0]=81
        g1[3] = (POS_PER_BLK << 16);   // tensor_dim1[31:16]=0 | tile_dim0 = 32
        g1[4] = OFFCH;                 // tile_dim1 = 81 | tile_dim2 = 0
        g1[5] = SPW;                   // tensor_dim0_stride[31:0] = 25088
        g1[6] = 0;                     // stride[47:32]=0 | tensor_dim1_stride lo
        g1[7] = 0;

        const v4i gz4 = {0, 0, 0, 0};            // groups 2/3 unused (2D tensor)
        const v8i gz8 = {0, 0, 0, 0, 0, 0, 0, 0};
        // 6-arg form (clang-23 / therock-10.0 headers): g0,g1,g2,g3,g4,cpol
        __builtin_amdgcn_tensor_load_to_lds(g0, g1, gz4, gz4, gz8, 0);
        __builtin_amdgcn_s_wait_tensorcnt(0);
    }
    __syncthreads();                   // publish offLDS to all waves

    // ------- Phase 1: trilinear im2col -> colLDS (f16) ---------------------
    for (int pair = tid; pair < POS_PER_BLK * KTAP; pair += 256) {
        const int lp  = pair / KTAP;
        const int tap = pair - lp * KTAP;
        const int p   = p0 + lp;
        const int rem = p - nb0 * SPW;
        const int t   = rem / (HH * WW);
        const int r2  = rem - t * (HH * WW);
        const int h   = r2 / WW;
        const int w   = r2 - h * WW;
        const int kt  = tap / 9;
        const int kh  = (tap / 3) % 3;
        const int kw  = tap % 3;

        // offsets from the TDM-staged LDS tile: [channel][pos]
        const int och = tap * 3;
        const float pt = (float)(t - 1 + kt) + offLDS[(och + 0) * POS_PER_BLK + lp];
        const float ph = (float)(h - 1 + kh) + offLDS[(och + 1) * POS_PER_BLK + lp];
        const float pw = (float)(w - 1 + kw) + offLDS[(och + 2) * POS_PER_BLK + lp];

        const float tf = floorf(pt), hf = floorf(ph), wf = floorf(pw);
        const int   t0 = (int)tf,    h0 = (int)hf,    w0 = (int)wf;
        const float lt = pt - tf,    lh = ph - hf,    lw = pw - wf;

        int   cidx[8];
        float cwt[8];
        #pragma unroll
        for (int i = 0; i < 8; ++i) {
            const int ct = i >> 2, ch = (i >> 1) & 1, cw2 = i & 1;
            const int ti = t0 + ct, hi = h0 + ch, wi = w0 + cw2;
            const bool valid = (ti >= 0) & (ti < TT) & (hi >= 0) & (hi < HH) &
                               (wi >= 0) & (wi < WW);
            const int tc = min(max(ti, 0), TT - 1);
            const int hc = min(max(hi, 0), HH - 1);
            const int wc = min(max(wi, 0), WW - 1);
            cidx[i] = (tc * HH + hc) * WW + wc;
            const float wv = (ct ? lt : 1.0f - lt) *
                             (ch ? lh : 1.0f - lh) *
                             (cw2 ? lw : 1.0f - lw);
            cwt[i] = valid ? wv : 0.0f;
        }

        const float* xb  = x + (size_t)nb0 * CIN * SPW;
        _Float16*    dst = &colLDS[lp * KDIM + tap];      // K index = c*27 + tap
        for (int c = 0; c < CIN; ++c) {
            const float* xc = xb + (size_t)c * SPW;
            float acc = 0.0f;
            #pragma unroll
            for (int i = 0; i < 8; ++i) acc += xc[cidx[i]] * cwt[i];
            dst[c * KTAP] = (_Float16)acc;
        }
    }
    __syncthreads();

    // ------- Phase 2: WMMA GEMM  D[64 x 32] = W[64 x 864] * col ------------
    const int mbase = (wave & 3) * 16;
    const int nbase = (wave >> 2) * 16;

    const int mrow  = mbase + (lane & 15);   // A-matrix row for this lane
    const int khalf = (lane >> 4) * 8;       // half-wave K sub-offset (ISA layout)
    const int npos  = nbase + (lane & 15);   // B-matrix column (local position)

    v8f acc = {};
    for (int kk = 0; kk < KDIM; kk += 32) {
        // A fragment: weights from global (f32 -> f16), 16-bit 16x32 layout
        v16h a;
        const float* wp = wgt + (size_t)mrow * KDIM + kk + khalf;
        #pragma unroll
        for (int v = 0; v < 8; ++v) {
            const int ko = (v < 4) ? (2 * v) : (16 + 2 * (v - 4));
            const float2 f2 = *(const float2*)(wp + ko);   // 8B aligned
            a[2 * v]     = (_Float16)f2.x;
            a[2 * v + 1] = (_Float16)f2.y;
        }
        // B fragment: f16 pairs from LDS col tile (4B-aligned dword reads)
        v8u bu;
        const _Float16* bp = &colLDS[npos * KDIM + kk + khalf];
        #pragma unroll
        for (int v = 0; v < 8; ++v) {
            const int ko = (v < 4) ? (2 * v) : (16 + 2 * (v - 4));
            bu[v] = *(const unsigned int*)(bp + ko);
        }
        const v16h b = __builtin_bit_cast(v16h, bu);

        // (neg_a, A, neg_b, B, c_mod, C, reuse_a, reuse_b)
        acc = __builtin_amdgcn_wmma_f32_16x16x32_f16(
            false, a, false, b, (short)0, acc, false, false);
    }

    // ------- Writeback: D layout (N = lane&15, M = r + 8*(lane>=16)) -------
    #pragma unroll
    for (int r = 0; r < 8; ++r) {
        const int m   = mbase + r + ((lane >> 4) << 3);
        const int p   = p0 + nbase + (lane & 15);
        const int rem = p - nb0 * SPW;
        out[((size_t)nb0 * COUT + m) * SPW + rem] = acc[r] + bias[m];
    }
}

extern "C" void kernel_launch(void* const* d_in, const int* in_sizes, int n_in,
                              void* d_out, int out_size, void* d_ws, size_t ws_size,
                              hipStream_t stream) {
    const float* x    = (const float*)d_in[0];   // [2, 32, 8, 56, 56]
    const float* off  = (const float*)d_in[1];   // [2, 81, 8, 56, 56]
    const float* wgt  = (const float*)d_in[2];   // [64, 32, 3, 3, 3]
    const float* bias = (const float*)d_in[3];   // [64]
    float*       out  = (float*)d_out;           // [2, 64, 8, 56, 56]

    (void)in_sizes; (void)n_in; (void)out_size; (void)d_ws; (void)ws_size;

    const int blocks = NPOS / POS_PER_BLK;       // 50176 / 32 = 1568, exact
    deform_conv3d_wmma<<<blocks, 256, 0, stream>>>(x, off, wgt, bias, out);
}
